// LinearGatedAttention_25048249270835
// MI455X (gfx1250) — matrix operand order repaired and checked by reading, not compiled, not run on hardware
//
#include <hip/hip_runtime.h>
#include <hip/hip_bf16.h>

// ---------------------------------------------------------------------------
// LinearGatedAttention for MI455X (gfx1250), bf16 WMMA pipeline.
// B=4, N=4096, DIM=1024, H=16, HD=64.
// q,k are stored head-transposed [b,h,hd,token] (token contiguous) so the
// q^T k WMMA fragments are contiguous vector loads instead of scalar gathers.
// GEMM B-tiles are staged with double-buffered GLOBAL_LOAD_ASYNC_TO_LDS_B128
// (ASYNCcnt), overlapping the DMA of tile t+1 with the WMMAs of tile t.
// ---------------------------------------------------------------------------

typedef __attribute__((ext_vector_type(16))) __bf16 v16bf;
typedef __attribute__((ext_vector_type(8)))  __bf16 v8bf;
typedef __attribute__((ext_vector_type(8)))  float  v8f;
typedef __attribute__((ext_vector_type(4)))  int    v4i;

#define B_    4
#define N_    4096
#define DIM_  1024
#define H_    16
#define HD_   64
#define M_    (B_ * N_)          // 16384 rows for the big GEMMs
#define SCALE_ 0.125f            // HD^-0.5

#define GLOBAL_AS __attribute__((address_space(1)))
#define LDS_AS    __attribute__((address_space(3)))

#if __has_builtin(__builtin_amdgcn_global_load_async_to_lds_b128) && \
    __has_builtin(__builtin_amdgcn_s_wait_asynccnt)
#define USE_ASYNC_LDS 1
#else
#define USE_ASYNC_LDS 0
#endif

#if USE_ASYNC_LDS
// Signature (from hipcc diagnostic): (v4i addrspace(1)* gsrc, v4i addrspace(3)* ldst,
//                                     imm int offset, imm int cpol)
__device__ __forceinline__ void async_copy_b128(const __bf16* gsrc, __bf16* ldst) {
  __builtin_amdgcn_global_load_async_to_lds_b128(
      (GLOBAL_AS v4i*)(v4i*)const_cast<__bf16*>(gsrc),
      (LDS_AS v4i*)(v4i*)ldst, 0, 0);
}
#endif

__device__ __forceinline__ float gelu_f(float x) {
  return 0.5f * x * (1.0f + erff(x * 0.70710678118654752440f));
}
__device__ __forceinline__ float softplus_f(float x) {
  return fmaxf(x, 0.0f) + log1pf(expf(-fabsf(x)));
}

// ---------------------------------------------------------------------------
// fp32 -> bf16 conversion (x and the 7 weight matrices)
// ---------------------------------------------------------------------------
__global__ __launch_bounds__(256) void cvt_f32_bf16(const float* __restrict__ s,
                                                    __bf16* __restrict__ d, int n) {
  int i = (blockIdx.x * 256 + threadIdx.x) * 4;
  if (i + 3 < n) {
    float4 f = *reinterpret_cast<const float4*>(s + i);
    d[i + 0] = (__bf16)f.x;
    d[i + 1] = (__bf16)f.y;
    d[i + 2] = (__bf16)f.z;
    d[i + 3] = (__bf16)f.w;
  }
}

// ---------------------------------------------------------------------------
// C = act(A[M,K] @ W[Nout,K]^T + bias)
// Block: 256 threads = 8 waves, tile 128(M) x 64(N), 4 wmma accums per wave.
// ACT: 0=none, 1=gelu, 2=softplus.
// LAYOUT: 0 = row-major [row, col]
//         1 = head-transposed [(b*H+h)*HD + n][token]  (token contiguous)
// ---------------------------------------------------------------------------
template <int ACT, int LAYOUT, typename OUT_T>
__global__ __launch_bounds__(256) void gemm_bias_act(
    const __bf16* __restrict__ A, const __bf16* __restrict__ W,
    const float* __restrict__ bias, OUT_T* __restrict__ C,
    int M, int K, int Nout) {
  __shared__ __align__(32) __bf16 sB[2][64 * 32];   // double-buffered [n=64][k=32]

  const int lane  = threadIdx.x & 31;
  const int wave  = threadIdx.x >> 5;
  const int colT  = lane & 15;
  const int hf    = lane >> 4;
  const int n_base = blockIdx.x * 64;
  const int m_base = blockIdx.y * 128 + wave * 16;

  v8f acc[4] = {v8f{}, v8f{}, v8f{}, v8f{}};

  const __bf16* __restrict__ arow = A + (size_t)(m_base + colT) * K;

  // Per-thread staging slot: thread t covers 16B: n = t/4, k-chunk = (t%4)*8
  const int stage_n  = threadIdx.x >> 2;
  const int stage_kc = (threadIdx.x & 3) * 8;
  const int stage_ofs = stage_n * 32 + stage_kc;
  const __bf16* __restrict__ wsrc =
      W + (size_t)(n_base + stage_n) * K + stage_kc;

#if USE_ASYNC_LDS
  // ---- async double-buffered pipeline: DMA(t+1) overlaps WMMA(t) ----
  async_copy_b128(wsrc, &sB[0][stage_ofs]);
  const int T = K / 32;
  for (int t = 0; t < T; ++t) {
    const int kk = t * 32;
    __builtin_amdgcn_s_wait_asynccnt(0);   // tile t landed (in-order per thread)
    __syncthreads();                       // visible block-wide; prior readers done
    if (t + 1 < T)
      async_copy_b128(wsrc + (size_t)(t + 1) * 32,
                      &sB[(t + 1) & 1][stage_ofs]);

    __builtin_prefetch(arow + kk + 128, 0, 1);
    const v8bf alo = *reinterpret_cast<const v8bf*>(arow + kk + hf * 16);
    const v8bf ahi = *reinterpret_cast<const v8bf*>(arow + kk + hf * 16 + 8);
    const v16bf afrag = __builtin_shufflevector(
        alo, ahi, 0, 1, 2, 3, 4, 5, 6, 7, 8, 9, 10, 11, 12, 13, 14, 15);

    const __bf16* sb = &sB[t & 1][0];
#pragma unroll
    for (int j = 0; j < 4; ++j) {
      const v16bf bfrag =
          *reinterpret_cast<const v16bf*>(&sb[(j * 16 + colT) * 32 + hf * 16]);
      acc[j] = __builtin_amdgcn_wmma_f32_16x16x32_bf16(
          false, afrag, false, bfrag, (short)0, acc[j], false, false);
    }
  }
#else
  // ---- fallback: synchronous staging, single buffer ----
  for (int kk = 0; kk < K; kk += 32) {
    {
      const uint4 v = *reinterpret_cast<const uint4*>(wsrc + kk);
      *reinterpret_cast<uint4*>(&sB[0][stage_ofs]) = v;
    }
    __syncthreads();

    __builtin_prefetch(arow + kk + 128, 0, 1);
    const v8bf alo = *reinterpret_cast<const v8bf*>(arow + kk + hf * 16);
    const v8bf ahi = *reinterpret_cast<const v8bf*>(arow + kk + hf * 16 + 8);
    const v16bf afrag = __builtin_shufflevector(
        alo, ahi, 0, 1, 2, 3, 4, 5, 6, 7, 8, 9, 10, 11, 12, 13, 14, 15);

#pragma unroll
    for (int j = 0; j < 4; ++j) {
      const v16bf bfrag = *reinterpret_cast<const v16bf*>(
          &sB[0][(j * 16 + colT) * 32 + hf * 16]);
      acc[j] = __builtin_amdgcn_wmma_f32_16x16x32_bf16(
          false, afrag, false, bfrag, (short)0, acc[j], false, false);
    }
    __syncthreads();
  }
#endif

  if (LAYOUT == 0) {
    // D layout: col = lane&15, rows = (lane>>4)*8 + r
#pragma unroll
    for (int j = 0; j < 4; ++j) {
      const int col = n_base + j * 16 + colT;
      const float bv = bias[col];
#pragma unroll
      for (int r = 0; r < 8; ++r) {
        const int row = m_base + hf * 8 + r;
        float v = acc[j][r] + bv;
        if (ACT == 1) v = gelu_f(v);
        else if (ACT == 2) v = softplus_f(v);
        C[(size_t)row * Nout + col] = (OUT_T)v;
      }
    }
  } else {
    // Head-transposed store: lane's 8 rows are 8 consecutive tokens ->
    // one contiguous 16B vector store per tile.
    const int row0 = m_base + hf * 8;      // global token row
    const int bb   = row0 >> 12;           // batch
    const int cc   = row0 & (N_ - 1);      // token within batch
#pragma unroll
    for (int j = 0; j < 4; ++j) {
      const int col = n_base + j * 16 + colT;   // feature = h*64 + n
      const float bv = bias[col];
      v8bf pack;
#pragma unroll
      for (int r = 0; r < 8; ++r) {
        float v = acc[j][r] + bv;
        if (ACT == 1) v = gelu_f(v);
        else if (ACT == 2) v = softplus_f(v);
        pack[r] = (__bf16)v;
      }
      __bf16* dst = (__bf16*)C + ((size_t)bb * DIM_ + col) * (size_t)N_ + cc;
      *reinterpret_cast<v8bf*>(dst) = pack;
    }
  }
}

// ---------------------------------------------------------------------------
// A[b,h,n,m] = sum_c qT[b,h,n,c] * kT[b,h,m,c]  (64x64 per head, K=4096 tokens)
// Block = 128 threads = 4 waves; each wave reduces 1024 tokens into a 4x4
// grid of 16x16 WMMA accumulators; partials combined through LDS (f32).
// ---------------------------------------------------------------------------
__global__ __launch_bounds__(128) void attn_A_kernel(const __bf16* __restrict__ qT,
                                                     const __bf16* __restrict__ kT,
                                                     __bf16* __restrict__ Abf) {
  __shared__ float sAcc[3 * HD_ * HD_];   // partials of waves 1..3 (48 KB)

  const int bh   = blockIdx.x;
  const int lane = threadIdx.x & 31;
  const int wave = threadIdx.x >> 5;
  const int colT = lane & 15;
  const int hf   = lane >> 4;

  const __bf16* __restrict__ qb = qT + (size_t)bh * HD_ * N_;  // [n][c]
  const __bf16* __restrict__ kb = kT + (size_t)bh * HD_ * N_;  // [m][c]

  v8f acc[16];
#pragma unroll
  for (int t = 0; t < 16; ++t) acc[t] = v8f{};

  const int cbeg = wave * (N_ / 4);
  for (int c0 = cbeg; c0 < cbeg + N_ / 4; c0 += 32) {
    v16bf qf[4], kf[4];
#pragma unroll
    for (int i = 0; i < 4; ++i) {
      const __bf16* p = qb + (size_t)(i * 16 + colT) * N_ + c0;
      const v8bf lo = *reinterpret_cast<const v8bf*>(p + hf * 16);
      const v8bf hi = *reinterpret_cast<const v8bf*>(p + hf * 16 + 8);
      qf[i] = __builtin_shufflevector(lo, hi, 0, 1, 2, 3, 4, 5, 6, 7, 8, 9, 10,
                                      11, 12, 13, 14, 15);
    }
#pragma unroll
    for (int j = 0; j < 4; ++j) {
      const __bf16* p = kb + (size_t)(j * 16 + colT) * N_ + c0;
      kf[j] = *reinterpret_cast<const v16bf*>(p + hf * 16);
    }
#pragma unroll
    for (int i = 0; i < 4; ++i)
#pragma unroll
      for (int j = 0; j < 4; ++j)
        acc[i * 4 + j] = __builtin_amdgcn_wmma_f32_16x16x32_bf16(
            false, qf[i], false, kf[j], (short)0, acc[i * 4 + j], false, false);
  }

  if (wave != 0) {
    float* dst = &sAcc[(wave - 1) * (HD_ * HD_)];
#pragma unroll
    for (int i = 0; i < 4; ++i)
#pragma unroll
      for (int j = 0; j < 4; ++j)
#pragma unroll
        for (int r = 0; r < 8; ++r)
          dst[(i * 16 + hf * 8 + r) * HD_ + j * 16 + colT] = acc[i * 4 + j][r];
  }
  __syncthreads();
  if (wave == 0) {
    __bf16* __restrict__ out = Abf + (size_t)bh * (HD_ * HD_);
#pragma unroll
    for (int i = 0; i < 4; ++i)
#pragma unroll
      for (int j = 0; j < 4; ++j)
#pragma unroll
        for (int r = 0; r < 8; ++r) {
          const int idx = (i * 16 + hf * 8 + r) * HD_ + j * 16 + colT;
          const float v = acc[i * 4 + j][r] + sAcc[idx] +
                          sAcc[HD_ * HD_ + idx] + sAcc[2 * HD_ * HD_ + idx];
          out[idx] = (__bf16)v;
        }
  }
}

// ---------------------------------------------------------------------------
// ks[b,h,c] = sum_m kT[b,h,m,c]   (coalesced across the token dim)
// ---------------------------------------------------------------------------
__global__ __launch_bounds__(256) void ks_kernel(const __bf16* __restrict__ kT,
                                                 float* __restrict__ ks) {
  const int bh = blockIdx.x >> 4;                       // grid = B*H*16
  const int c  = (blockIdx.x & 15) * 256 + threadIdx.x; // token
  const __bf16* __restrict__ p = kT + (size_t)bh * HD_ * N_ + c;
  float s = 0.0f;
#pragma unroll
  for (int m = 0; m < HD_; ++m) s += (float)p[(size_t)m * N_];
  ks[(size_t)bh * N_ + c] = s;
}

// ---------------------------------------------------------------------------
// z[b,h,n] = 1 / (SCALE * sum_c qT[b,h,n,c]*ks[b,h,c] + DIM)
// ---------------------------------------------------------------------------
__global__ __launch_bounds__(64) void z_kernel(const __bf16* __restrict__ qT,
                                               const float* __restrict__ ks,
                                               float* __restrict__ z) {
  const int bh = blockIdx.x;
  const int n  = threadIdx.x;  // 0..63
  const __bf16* __restrict__ q = qT + ((size_t)bh * HD_ + n) * N_;
  const float* __restrict__ kp = ks + (size_t)bh * N_;
  float acc = 0.0f;
  for (int c = 0; c < N_; c += 8) {
    const v8bf qv = *reinterpret_cast<const v8bf*>(q + c);
#pragma unroll
    for (int e = 0; e < 8; ++e) acc += (float)qv[e] * kp[c + e];
  }
  z[bh * HD_ + n] = 1.0f / (SCALE_ * acc + (float)DIM_);
}

// ---------------------------------------------------------------------------
// out[t,n] = SCALE * sum_m v[t,m]*A[n,m];  y = (out + v[t,n]) * z[n] * g[t,n]
// Stored flat as [b][h][t][n] -> raw reshape to [B,N,DIM] for the projection.
// ---------------------------------------------------------------------------
__global__ __launch_bounds__(256) void out_kernel(
    const __bf16* __restrict__ V, const __bf16* __restrict__ G,
    const __bf16* __restrict__ Abf, const float* __restrict__ z,
    __bf16* __restrict__ Y) {
  const int bh = blockIdx.x;
  const int b  = bh >> 4;
  const int h  = bh & 15;
  const int lane = threadIdx.x & 31;
  const int wave = threadIdx.x >> 5;
  const int colT = lane & 15;
  const int hf   = lane >> 4;
  const int t0   = blockIdx.y * 128 + wave * 16;

  v8f acc[4] = {v8f{}, v8f{}, v8f{}, v8f{}};

  const __bf16* __restrict__ vrow =
      V + ((size_t)(b * N_ + t0 + colT)) * DIM_ + h * HD_;
  const __bf16* __restrict__ Ah = Abf + (size_t)bh * (HD_ * HD_);

#pragma unroll
  for (int m0 = 0; m0 < HD_; m0 += 32) {
    const v8bf alo = *reinterpret_cast<const v8bf*>(vrow + m0 + hf * 16);
    const v8bf ahi = *reinterpret_cast<const v8bf*>(vrow + m0 + hf * 16 + 8);
    const v16bf vfrag = __builtin_shufflevector(
        alo, ahi, 0, 1, 2, 3, 4, 5, 6, 7, 8, 9, 10, 11, 12, 13, 14, 15);
#pragma unroll
    for (int j = 0; j < 4; ++j) {
      const v16bf bfrag = *reinterpret_cast<const v16bf*>(
          Ah + (j * 16 + colT) * HD_ + m0 + hf * 16);
      acc[j] = __builtin_amdgcn_wmma_f32_16x16x32_bf16(
          false, vfrag, false, bfrag, (short)0, acc[j], false, false);
    }
  }

#pragma unroll
  for (int j = 0; j < 4; ++j) {
    const int n = j * 16 + colT;
    const float zv = z[bh * HD_ + n];
#pragma unroll
    for (int r = 0; r < 8; ++r) {
      const int t = t0 + hf * 8 + r;
      const size_t idx = ((size_t)(b * N_ + t)) * DIM_ + h * HD_ + n;
      const float o  = SCALE_ * acc[j][r];
      const float vv = (float)V[idx];
      const float gg = (float)G[idx];
      const float y  = (o + vv) * zv * gg;
      Y[(size_t)b * N_ * DIM_ + ((size_t)h * N_ + t) * HD_ + n] = (__bf16)y;
    }
  }
}

// ---------------------------------------------------------------------------
// Host-side orchestration.
// Inputs: x, q_w1, q_b1, q_w2, q_b2, k_w1, k_b1, k_w2, k_b2,
//         v_w, v_b, g_w, g_b, p_w, p_b   (all fp32)
// ---------------------------------------------------------------------------
extern "C" void kernel_launch(void* const* d_in, const int* in_sizes, int n_in,
                              void* d_out, int out_size, void* d_ws, size_t ws_size,
                              hipStream_t stream) {
  (void)in_sizes; (void)n_in; (void)out_size; (void)ws_size;

  const float* x   = (const float*)d_in[0];
  const float* qb1 = (const float*)d_in[2];
  const float* qb2 = (const float*)d_in[4];
  const float* kb1 = (const float*)d_in[6];
  const float* kb2 = (const float*)d_in[8];
  const float* vb  = (const float*)d_in[10];
  const float* gb  = (const float*)d_in[12];
  const float* pb  = (const float*)d_in[14];

  // Workspace layout (bytes)
  const size_t ACT  = (size_t)M_ * DIM_ * sizeof(__bf16);  // 32 MiB
  const size_t WSZ  = (size_t)DIM_ * DIM_ * sizeof(__bf16);
  char* ws = (char*)d_ws;
  __bf16* buf0 = (__bf16*)(ws + 0 * ACT);   // xb, later qT
  __bf16* buf1 = (__bf16*)(ws + 1 * ACT);   // hq, later kT
  __bf16* buf2 = (__bf16*)(ws + 2 * ACT);   // hk, later y2 (gated output)
  __bf16* buf3 = (__bf16*)(ws + 3 * ACT);   // v
  __bf16* buf4 = (__bf16*)(ws + 4 * ACT);   // g
  __bf16* wb   = (__bf16*)(ws + 5 * ACT);   // 7 bf16 weight matrices
  char*   tail = ws + 5 * ACT + 7 * WSZ;
  __bf16* Abf  = (__bf16*)tail;                                   // 64*64*64 bf16
  float*  ksb  = (float*)(tail + (size_t)B_ * H_ * HD_ * HD_ * 2);// B*H*N f32
  float*  zb   = (float*)((char*)ksb + (size_t)B_ * H_ * N_ * 4); // B*H*HD f32

  __bf16* w_q1 = wb + 0 * (size_t)DIM_ * DIM_;
  __bf16* w_q2 = wb + 1 * (size_t)DIM_ * DIM_;
  __bf16* w_k1 = wb + 2 * (size_t)DIM_ * DIM_;
  __bf16* w_k2 = wb + 3 * (size_t)DIM_ * DIM_;
  __bf16* w_v  = wb + 4 * (size_t)DIM_ * DIM_;
  __bf16* w_g  = wb + 5 * (size_t)DIM_ * DIM_;
  __bf16* w_p  = wb + 6 * (size_t)DIM_ * DIM_;

  // --- 1. convert x and weights to bf16 ---
  {
    const int nx = M_ * DIM_;
    cvt_f32_bf16<<<nx / (256 * 4), 256, 0, stream>>>(x, buf0, nx);
    const int nw = DIM_ * DIM_;
    const int widx[7] = {1, 3, 5, 7, 9, 11, 13};
    __bf16* wdst[7] = {w_q1, w_q2, w_k1, w_k2, w_v, w_g, w_p};
    for (int i = 0; i < 7; ++i)
      cvt_f32_bf16<<<nw / (256 * 4), 256, 0, stream>>>(
          (const float*)d_in[widx[i]], wdst[i], nw);
  }

  const dim3 ggrid(DIM_ / 64, M_ / 128);  // (16, 128)

  // --- 2. first-layer GEMMs (gelu), all read xb = buf0 ---
  gemm_bias_act<1, 0, __bf16><<<ggrid, 256, 0, stream>>>(buf0, w_q1, qb1, buf1, M_, DIM_, DIM_); // hq
  gemm_bias_act<1, 0, __bf16><<<ggrid, 256, 0, stream>>>(buf0, w_k1, kb1, buf2, M_, DIM_, DIM_); // hk
  gemm_bias_act<1, 0, __bf16><<<ggrid, 256, 0, stream>>>(buf0, w_v,  vb,  buf3, M_, DIM_, DIM_); // v
  gemm_bias_act<1, 0, __bf16><<<ggrid, 256, 0, stream>>>(buf0, w_g,  gb,  buf4, M_, DIM_, DIM_); // g

  // --- 3. second-layer GEMMs (softplus), head-transposed outputs ---
  gemm_bias_act<2, 1, __bf16><<<ggrid, 256, 0, stream>>>(buf1, w_q2, qb2, buf0, M_, DIM_, DIM_); // qT
  gemm_bias_act<2, 1, __bf16><<<ggrid, 256, 0, stream>>>(buf2, w_k2, kb2, buf1, M_, DIM_, DIM_); // kT

  // --- 4. per-head reductions ---
  attn_A_kernel<<<B_ * H_, 128, 0, stream>>>(buf0, buf1, Abf);
  ks_kernel<<<B_ * H_ * 16, 256, 0, stream>>>(buf1, ksb);
  z_kernel<<<B_ * H_, 64, 0, stream>>>(buf0, ksb, zb);

  // --- 5. gated output, stored raw-reshaped (over hk buffer) ---
  out_kernel<<<dim3(B_ * H_, N_ / 128), 256, 0, stream>>>(buf3, buf4, Abf, zb, buf2);

  // --- 6. final projection to fp32 output ---
  gemm_bias_act<0, 0, float><<<ggrid, 256, 0, stream>>>(buf2, w_p, pb, (float*)d_out, M_, DIM_, DIM_);
}
